// CapsuleLayer_81913616270182
// MI455X (gfx1250) — matrix operand order repaired and checked by reading, not compiled, not run on hardware
//
#include <hip/hip_runtime.h>
#include <hip/hip_bf16.h>
#include <math.h>

// Problem constants (from reference setup_inputs)
#define BSZ 64
#define ICAP 2048
#define JCAP 32
#define DD 128
#define VV 32
#define NCOL 1024 // J*V

typedef __attribute__((ext_vector_type(16))) __bf16 v16bf;
typedef __attribute__((ext_vector_type(8)))  float  v8f;

// ---------------------------------------------------------------------------
// Kernel 1: u_hat[b,i,j,v] = sum_d W[i,j,d,v] * x[b,i,d]   (bf16 WMMA, f32 acc)
// Grid: (ICAP, 4). Block: 256 (8 waves). Each WG: one capsule i, 256 columns.
// Wave w owns n-tiles {chunk*16 + 2w, +1}, all 4 M(batch)-tiles.
// ---------------------------------------------------------------------------
__global__ __launch_bounds__(256)
void gemm_uhat(const float* __restrict__ x, const float* __restrict__ W,
               __bf16* __restrict__ uhat) {
  const int i     = blockIdx.x;
  const int chunk = blockIdx.y;
  const int wave  = threadIdx.x >> 5;
  const int lane  = threadIdx.x & 31;
  const int grp   = lane >> 4;   // half-wave group
  const int l16   = lane & 15;
  const int nt0   = chunk * 16 + wave * 2;

  v8f acc[4][2];
#pragma unroll
  for (int mt = 0; mt < 4; ++mt)
#pragma unroll
    for (int t = 0; t < 2; ++t) acc[mt][t] = (v8f)0.f;

  for (int kk = 0; kk < 4; ++kk) {
    const int kbase = kk * 32;

    // ---- A fragments (x), ISA 16-bit A 16x32 layout:
    // lane grp 0 holds K = {0..7, 16..23}+kbase ; grp 1 holds {8..15, 24..31}
    v16bf a[4];
#pragma unroll
    for (int mt = 0; mt < 4; ++mt) {
      const int b = mt * 16 + l16;
      const float* p = x + ((size_t)b * ICAP + i) * DD + kbase + grp * 8;
      float4 r0 = ((const float4*)p)[0];
      float4 r1 = ((const float4*)p)[1];
      float4 r2 = ((const float4*)(p + 16))[0];
      float4 r3 = ((const float4*)(p + 16))[1];
      a[mt][0]  = (__bf16)r0.x; a[mt][1]  = (__bf16)r0.y;
      a[mt][2]  = (__bf16)r0.z; a[mt][3]  = (__bf16)r0.w;
      a[mt][4]  = (__bf16)r1.x; a[mt][5]  = (__bf16)r1.y;
      a[mt][6]  = (__bf16)r1.z; a[mt][7]  = (__bf16)r1.w;
      a[mt][8]  = (__bf16)r2.x; a[mt][9]  = (__bf16)r2.y;
      a[mt][10] = (__bf16)r2.z; a[mt][11] = (__bf16)r2.w;
      a[mt][12] = (__bf16)r3.x; a[mt][13] = (__bf16)r3.y;
      a[mt][14] = (__bf16)r3.z; a[mt][15] = (__bf16)r3.w;
    }

    // ---- B fragments (W) + WMMA. B 32x16: lane grp 0 -> K {0..7,16..23}, grp 1 -> K {8..15,24..31}
#pragma unroll
    for (int t = 0; t < 2; ++t) {
      const int nt  = nt0 + t;
      const int col = nt * 16 + l16;       // col = j*32 + v; 16-tile stays in one j
      const int j = col >> 5, v = col & 31;
      const float* q =
          W + (((size_t)i * JCAP + j) * DD + (kbase + grp * 8)) * VV + v;
      v16bf bf;
#pragma unroll
      for (int h = 0; h < 16; ++h) bf[h] = (__bf16)q[(size_t)((h & 7) + (h >> 3) * 16) * VV];

#pragma unroll
      for (int mt = 0; mt < 4; ++mt)
        acc[mt][t] = __builtin_amdgcn_wmma_f32_16x16x32_bf16(
            false, a[mt], false, bf, (short)0, acc[mt][t], false, false);
    }
  }

  // ---- Store C: VGPR r -> row M = r + 8*grp, col = lane&15 (ISA C/D layout)
#pragma unroll
  for (int mt = 0; mt < 4; ++mt)
#pragma unroll
    for (int t = 0; t < 2; ++t) {
      const int col = (nt0 + t) * 16 + l16;
#pragma unroll
      for (int r = 0; r < 8; ++r) {
        const int b = mt * 16 + r + grp * 8;
        uhat[((size_t)b * ICAP + i) * NCOL + col] = (__bf16)acc[mt][t][r];
      }
    }
}

// ---------------------------------------------------------------------------
// Kernel 2: s[b,j,v] = sum_i c[i,j] * u_hat[b,i,j,v]   (c==nullptr -> uniform)
// Grid: B*J blocks. 8 waves split I; lanes = v (coalesced 64B bf16 rows).
// ---------------------------------------------------------------------------
__global__ __launch_bounds__(256)
void reduce_s(const __bf16* __restrict__ uhat, const float* __restrict__ c,
              float cunif, float* __restrict__ s) {
  const int bj = blockIdx.x;
  const int b = bj >> 5, j = bj & 31;
  const int wave = threadIdx.x >> 5, lane = threadIdx.x & 31;
  float acc = 0.f;
  for (int i = wave; i < ICAP; i += 8) {
    const float cc = c ? c[i * JCAP + j] : cunif;
    const float u = (float)uhat[(((size_t)b * ICAP + i) * JCAP + j) * VV + lane];
    acc += cc * u;
  }
  __shared__ float red[8][32];
  red[wave][lane] = acc;
  __syncthreads();
  if (wave == 0) {
    float t = 0.f;
#pragma unroll
    for (int w2 = 0; w2 < 8; ++w2) t += red[w2][lane];
    s[(size_t)bj * VV + lane] = t;
  }
}

// ---------------------------------------------------------------------------
// Kernel 3: squash (faithful: reduces over J, dim=2 of [B,1,J,V])
// Grid: B blocks of 1024 threads; thread t -> (j = t>>5, v = t&31).
// ---------------------------------------------------------------------------
__global__ __launch_bounds__(1024)
void squash_k(const float* __restrict__ s, float* __restrict__ out) {
  const int b = blockIdx.x;
  const int t = threadIdx.x;
  const int v = t & 31;
  const float sv = s[(size_t)b * 1024 + t];
  __shared__ float sq[1024];
  sq[t] = sv * sv;
  __syncthreads();
  float mag_sq = 0.f;
#pragma unroll
  for (int j = 0; j < 32; ++j) mag_sq += sq[j * 32 + v];
  const float mag = sqrtf(mag_sq + 1e-9f);
  out[(size_t)b * 1024 + t] = mag_sq / (1.f + mag_sq) * (sv / mag);
}

// ---------------------------------------------------------------------------
// Kernel 4: b_ij[i,j] = (first?0:b_ij) + (1/B) * sum_{b,v} uhat[b,i,j,v]*vb[b,j,v]
// Grid: I blocks; wave w owns j = 4w..4w+3; lanes = v; shfl reduce over v.
// ---------------------------------------------------------------------------
__global__ __launch_bounds__(256)
void agree_update(const __bf16* __restrict__ uhat, const float* __restrict__ vb,
                  float* __restrict__ bij, int first) {
  const int i = blockIdx.x;
  const int wave = threadIdx.x >> 5, lane = threadIdx.x & 31;
  for (int jj = 0; jj < 4; ++jj) {
    const int j = wave * 4 + jj;
    float acc = 0.f;
    for (int b = 0; b < BSZ; ++b) {
      const float u =
          (float)uhat[(((size_t)b * ICAP + i) * JCAP + j) * VV + lane];
      acc += u * vb[((size_t)b * JCAP + j) * VV + lane];
    }
#pragma unroll
    for (int off = 16; off > 0; off >>= 1) acc += __shfl_xor(acc, off, 32);
    if (lane == 0) {
      const float old = first ? 0.f : bij[i * JCAP + j];
      bij[i * JCAP + j] = old + acc * (1.f / BSZ);
    }
  }
}

// ---------------------------------------------------------------------------
// Kernel 5: c[i,j] = softmax over i of b_ij[i,j].  Grid: J blocks.
// ---------------------------------------------------------------------------
__global__ __launch_bounds__(256)
void softmax_i(const float* __restrict__ bij, float* __restrict__ c) {
  const int j = blockIdx.x;
  const int t = threadIdx.x;
  __shared__ float red[256];
  float m = -1e30f;
  for (int i = t; i < ICAP; i += 256) m = fmaxf(m, bij[i * JCAP + j]);
  red[t] = m; __syncthreads();
  for (int s2 = 128; s2 > 0; s2 >>= 1) {
    if (t < s2) red[t] = fmaxf(red[t], red[t + s2]);
    __syncthreads();
  }
  m = red[0]; __syncthreads();
  float sum = 0.f;
  for (int i = t; i < ICAP; i += 256) sum += __expf(bij[i * JCAP + j] - m);
  red[t] = sum; __syncthreads();
  for (int s2 = 128; s2 > 0; s2 >>= 1) {
    if (t < s2) red[t] += red[t + s2];
    __syncthreads();
  }
  const float inv = 1.f / red[0];
  for (int i = t; i < ICAP; i += 256)
    c[i * JCAP + j] = __expf(bij[i * JCAP + j] - m) * inv;
}

// ---------------------------------------------------------------------------
extern "C" void kernel_launch(void* const* d_in, const int* in_sizes, int n_in,
                              void* d_out, int out_size, void* d_ws,
                              size_t ws_size, hipStream_t stream) {
  (void)in_sizes; (void)n_in; (void)out_size; (void)ws_size;
  const float* x = (const float*)d_in[0];
  const float* W = (const float*)d_in[1];
  float* out = (float*)d_out;

  char* ws = (char*)d_ws;
  __bf16* uhat = (__bf16*)ws;
  size_t off = (size_t)BSZ * ICAP * JCAP * VV * sizeof(__bf16); // 256 MiB
  float* s   = (float*)(ws + off); off += (size_t)BSZ * JCAP * VV * 4;
  float* vb  = (float*)(ws + off); off += (size_t)BSZ * JCAP * VV * 4;
  float* bij = (float*)(ws + off); off += (size_t)ICAP * JCAP * 4;
  float* c   = (float*)(ws + off);

  dim3 g1(ICAP, 4);
  gemm_uhat<<<g1, 256, 0, stream>>>(x, W, uhat);

  // iter 0: uniform coupling
  reduce_s<<<BSZ * JCAP, 256, 0, stream>>>(uhat, nullptr, 1.f / ICAP, s);
  squash_k<<<BSZ, 1024, 0, stream>>>(s, vb);
  agree_update<<<ICAP, 256, 0, stream>>>(uhat, vb, bij, 1);
  softmax_i<<<JCAP, 256, 0, stream>>>(bij, c);

  // iter 1
  reduce_s<<<BSZ * JCAP, 256, 0, stream>>>(uhat, c, 0.f, s);
  squash_k<<<BSZ, 1024, 0, stream>>>(s, vb);
  agree_update<<<ICAP, 256, 0, stream>>>(uhat, vb, bij, 0);
  softmax_i<<<JCAP, 256, 0, stream>>>(bij, c);

  // iter 2 (final): squash straight into d_out [B,J,V]
  reduce_s<<<BSZ * JCAP, 256, 0, stream>>>(uhat, c, 0.f, s);
  squash_k<<<BSZ, 1024, 0, stream>>>(s, out);
}
